// GRUNeuralHawkesProcess_3410204033608
// MI455X (gfx1250) — matrix operand order repaired and checked by reading, not compiled, not run on hardware
//
#include <hip/hip_runtime.h>
#include <hip/hip_bf16.h>

typedef __attribute__((ext_vector_type(16))) _Float16 v16h;
typedef __attribute__((ext_vector_type(8)))  _Float16 v8h;
typedef __attribute__((ext_vector_type(8)))  float    v8f;
typedef __attribute__((ext_vector_type(4)))  float    v4f;

#define B_   64
#define L_   2048
#define LP1  (L_ + 1)
#define H_   64
#define M_   4
#define BT   16          // batches per workgroup (== WMMA M)
#define NT   128         // 4 waves of 32
#define T_END_F 1000.0f
#define LN10 2.302585093f

// ---------------------------------------------------------------------------
// delta[b][j] = t_j - t_{j-1} (j<len), T_END - t_last (j==len), -1 else.
// ---------------------------------------------------------------------------
__global__ void build_delta_kernel(const float* __restrict__ seq_pads,
                                   const int*   __restrict__ seq_lens,
                                   float*       __restrict__ delta) {
  const int b  = blockIdx.x;
  const int sl = seq_lens[b];
  const float t_last = seq_pads[(size_t)b * L_ + (sl - 1)];
  for (int j = threadIdx.x; j <= L_; j += blockDim.x) {
    float v;
    if (j < sl) {
      float cur  = seq_pads[(size_t)b * L_ + j];
      float prev = (j > 0) ? seq_pads[(size_t)b * L_ + j - 1] : 0.0f;
      v = cur - prev;
    } else if (j == sl) {
      v = T_END_F - t_last;
    } else {
      v = -1.0f;
    }
    delta[(size_t)b * LP1 + j] = v;
  }
}

// ---------------------------------------------------------------------------
// Parallel epilogue: befores = afters * exp(-delta / tau_m).  Pure bandwidth
// (134 MB read + 134 MB write ~ 11.5 us at 23.3 TB/s); afters mostly L2-hot.
// ---------------------------------------------------------------------------
__global__ void befores_kernel(const float* __restrict__ afters,
                               const float* __restrict__ delta,
                               float*       __restrict__ befores) {
  const size_t total4 = (size_t)M_ * B_ * LP1 * (H_ / 4);
  const size_t i = (size_t)blockIdx.x * blockDim.x + threadIdx.x;
  if (i >= total4) return;
  const size_t row = i >> 4;                       // (m*B + b)*LP1 + j
  const int m = (int)(row / (size_t)(B_ * LP1));
  const size_t rem = row - (size_t)m * (B_ * LP1); // == b*LP1 + j
  const float d = delta[rem];
  const float inv = (m == 0) ? 1.0f : (m == 1) ? 0.1f
                  : (m == 2) ? 0.01f : 0.001f;
  const float e = __expf(-d * inv);
  const v4f a = *(const v4f*)(afters + 4 * i);
  v4f o;
  o.x = a.x * e; o.y = a.y * e; o.z = a.z * e; o.w = a.w * e;
  *(v4f*)(befores + 4 * i) = o;
}

// ---------------------------------------------------------------------------
// WMMA fragment helpers (16-bit operand layouts per CDNA5 ISA 7.12.2)
// ---------------------------------------------------------------------------
__device__ __forceinline__ v16h load_afrag(const _Float16* __restrict__ A,
                                           int kbase, int lane) {
  const int row  = lane & 15;
  const int koff = (lane < 16) ? 0 : 8;
  v8h lo = *(const v8h*)&A[row * H_ + kbase + koff];
  v8h hi = *(const v8h*)&A[row * H_ + kbase + 16 + koff];
  return __builtin_shufflevector(lo, hi, 0, 1, 2, 3, 4, 5, 6, 7,
                                 8, 9, 10, 11, 12, 13, 14, 15);
}

__device__ __forceinline__ v16h load_bfrag_g(const float* __restrict__ W,
                                             int nbase, int kbase, int lane) {
  const int n    = nbase + (lane & 15);
  const int koff = (lane < 16) ? 0 : 8;
  v16h b;
#pragma unroll
  for (int i = 0; i < 16; ++i)
    b[i] = (_Float16)W[(kbase + koff + (i < 8 ? i : i + 8)) * H_ + n];
  return b;
}

__device__ __forceinline__ v8f wmma_k64(v16h a0, v16h a1, v16h b0, v16h b1) {
  v8f c = {0.f, 0.f, 0.f, 0.f, 0.f, 0.f, 0.f, 0.f};
  c = __builtin_amdgcn_wmma_f32_16x16x32_f16(false, a0, false, b0,
                                             (short)0, c, false, false);
  c = __builtin_amdgcn_wmma_f32_16x16x32_f16(false, a1, false, b1,
                                             (short)0, c, false, false);
  return c;
}

__device__ __forceinline__ float fast_tanh(float x) {
  const float e = __expf(2.0f * x);
  return 1.0f - 2.0f * __builtin_amdgcn_rcpf(e + 1.0f);
}

// Unnormalized softmax weights over m of exp(-(m*LN10 - x)^2), ONE exp:
//   w_m ∝ D_m * e2^m,  e2 = exp(2*LN10*(x - 1.5*LN10)),
//   D0 = D3 = exp(-2.25*LN10^2), D1 = D2 = exp(-0.25*LN10^2).
// u clamped to ±6.3 so e2^3 stays finite; clamp only perturbs the ~1e-8
// saturated tail of the softmax (reference ratios preserved to ~1e-6).
__device__ __forceinline__ v4f softmax4_scales(float x) {
  float u = x - 1.5f * LN10;
  u = fminf(fmaxf(u, -6.3f), 6.3f);
  const float e2 = __expf(2.0f * LN10 * u);
  const float e4 = e2 * e2;
  const float D0 = 6.5942e-6f;      // exp(-2.25*LN10^2)
  const float D1 = 0.2656862f;      // exp(-0.25*LN10^2)
  v4f w;
  w.x = D0;
  w.y = D1 * e2;
  w.z = D1 * e4;
  w.w = D0 * (e4 * e2);
  return w;
}

// ---------------------------------------------------------------------------
// CTGRU scan. One WG owns 16 batches; 4 waves each own a 16-column N-tile.
// State hhat[8][4] lives in VGPRs (lane-local under the fixed WMMA C/D
// mapping b = v + 8*(lane>=16), h = nbase + (lane&15)). Only the WMMA A
// operands (hsum, hretr) cross waves through LDS. 2 barriers per step; the
// "befores" decay is hoisted to the parallel epilogue kernel.
// ---------------------------------------------------------------------------
__global__ __launch_bounds__(NT, 1) void ctgru_scan_kernel(
    const float* __restrict__ Wr, const float* __restrict__ Ur,
    const float* __restrict__ br, const float* __restrict__ Wq,
    const float* __restrict__ Uq, const float* __restrict__ bq,
    const float* __restrict__ Ws, const float* __restrict__ Us,
    const float* __restrict__ bs, const float* __restrict__ delta,
    float* __restrict__ afters) {
  __shared__ __align__(16) _Float16 s_hsum16[BT * H_];
  __shared__ __align__(16) _Float16 s_hretr16[BT * H_];
  __shared__ __align__(16) float s_Ebuf[2][BT][M_];  // exp(-dt/tau), b-major
  __shared__ float s_dtbuf[2][BT];

  const int tid   = threadIdx.x;
  const int lane  = tid & 31;
  const int wave  = tid >> 5;
  const int bb    = blockIdx.x * BT;
  const int nbase = wave * 16;
  const int hcol  = nbase + (lane & 15);   // my GEMM output column
  const int boff  = (lane < 16) ? 0 : 8;   // C/D row base per ISA layout

  // ---- one-time init: afters position 0 = h0 = 0 ----
  for (int i = tid; i < M_ * BT * H_; i += NT) {
    const int m = i >> 10, b = (i >> 6) & (BT - 1), h = i & (H_ - 1);
    afters[((size_t)(m * B_ + bb + b) * LP1 + 0) * H_ + h] = 0.0f;
  }
  if (tid < BT * M_) {          // prime decay buffer + dt for step 0
    const int b = tid >> 2, m = tid & 3;
    const float dt = delta[(size_t)(bb + b) * LP1 + 0];
    const float inv = (m == 0) ? 1.0f : (m == 1) ? 0.1f
                    : (m == 2) ? 0.01f : 0.001f;
    s_Ebuf[0][b][m] = __expf(-dt * inv);
    if (m == 0) s_dtbuf[0][b] = dt;
  }

  // persistent weight B-fragments (K = 64 -> two K=32 chunks each)
  const v16h fUr0 = load_bfrag_g(Ur, nbase, 0, lane);
  const v16h fUr1 = load_bfrag_g(Ur, nbase, 32, lane);
  const v16h fUq0 = load_bfrag_g(Uq, nbase, 0, lane);
  const v16h fUq1 = load_bfrag_g(Uq, nbase, 32, lane);
  const v16h fUs0 = load_bfrag_g(Us, nbase, 0, lane);
  const v16h fUs1 = load_bfrag_g(Us, nbase, 32, lane);

  // per-lane row-vector / bias scalars for my column
  const float wvr = Wr[hcol], wvq = Wq[hcol], wvs = Ws[hcol];
  const float bvr = br[hcol], bvq = bq[hcol], bvs = bs[hcol];

  // register-resident state and decayed state
  float hreg[8][M_];
  float dec[8][M_];
#pragma unroll
  for (int v = 0; v < 8; ++v)
#pragma unroll
    for (int m = 0; m < M_; ++m) hreg[v][m] = 0.0f;

  __syncthreads();   // s_Ebuf[0]/s_dtbuf[0] visible

  for (int t = 0; t < L_; ++t) {
    const int ct = t & 1, nt = (t + 1) & 1;

    // ---- phase A: decay + dt for step t+1. Ordered before its consumers
    // (B', C') by this step's barriers.
    if (tid < BT * M_) {
      const int b = tid >> 2, m = tid & 3;
      const float dtn = delta[(size_t)(bb + b) * LP1 + t + 1];
      const float inv = (m == 0) ? 1.0f : (m == 1) ? 0.1f
                      : (m == 2) ? 0.01f : 0.001f;
      s_Ebuf[nt][b][m] = __expf(-dtn * inv);
      if (m == 0) s_dtbuf[nt][b] = dtn;
    }

    // ---- phase B: decayed state + hsum (register state, wave-local) ----
#pragma unroll
    for (int v = 0; v < 8; ++v) {
      const int b = v + boff;
      const v4f e = *(const v4f*)&s_Ebuf[ct][b][0];
      float hs = 0.0f;
#pragma unroll
      for (int m = 0; m < M_; ++m) {
        dec[v][m] = hreg[v][m] * e[m];
        hs += dec[v][m];
      }
      s_hsum16[b * H_ + hcol] = (_Float16)hs;
    }
    __syncthreads();   // BARRIER 1

    // ---- phase C: ln_r/ln_s GEMMs (WMMA) + r-softmax + hretr ----
    float lns_v[8], dt_v[8];
    {
      const v16h a0 = load_afrag(s_hsum16, 0, lane);
      const v16h a1 = load_afrag(s_hsum16, 32, lane);
      const v8f cr = wmma_k64(a0, a1, fUr0, fUr1);
      const v8f cs = wmma_k64(a0, a1, fUs0, fUs1);
#pragma unroll
      for (int v = 0; v < 8; ++v) {
        const int b = v + boff;
        const float dtv = s_dtbuf[ct][b];
        dt_v[v] = dtv;                      // cached for phase D (regs)
        const float xr = cr[v] + dtv * wvr + bvr;
        lns_v[v] = cs[v] + dtv * wvs + bvs;
        const v4f w = softmax4_scales(xr);
        const float sum = (w.x + w.y) + (w.z + w.w);
        const float dot = w.x * dec[v][0] + w.y * dec[v][1] +
                          w.z * dec[v][2] + w.w * dec[v][3];
        const float hr = dot * __builtin_amdgcn_rcpf(sum);
        s_hretr16[b * H_ + hcol] = (_Float16)hr;
      }
    }
    __syncthreads();   // BARRIER 2

    // ---- phase D: q GEMM (WMMA) + s-gate + state update + afters stores.
    // Reads only s_hretr16 (+regs); all rewrites are behind next iteration's
    // barriers, so no end-of-step barrier is needed.
    {
      const v16h a0 = load_afrag(s_hretr16, 0, lane);
      const v16h a1 = load_afrag(s_hretr16, 32, lane);
      const v8f cq = wmma_k64(a0, a1, fUq0, fUq1);
#pragma unroll
      for (int v = 0; v < 8; ++v) {
        const int b = v + boff;
        const float q = fast_tanh(cq[v] + dt_v[v] * wvq + bvq);
        const v4f w = softmax4_scales(lns_v[v]);
        const float sum = (w.x + w.y) + (w.z + w.w);
        const float inv = __builtin_amdgcn_rcpf(sum);
#pragma unroll
        for (int m = 0; m < M_; ++m) {
          const float sm = w[m] * inv;
          const float hn = dec[v][m] + sm * (q - dec[v][m]);
          hreg[v][m] = hn;
          afters[((size_t)(m * B_ + bb + b) * LP1 + (t + 1)) * H_ + hcol] = hn;
        }
      }
    }
  }
}

// ---------------------------------------------------------------------------
extern "C" void kernel_launch(void* const* d_in, const int* in_sizes, int n_in,
                              void* d_out, int out_size, void* d_ws,
                              size_t ws_size, hipStream_t stream) {
  (void)in_sizes; (void)n_in; (void)out_size; (void)d_ws; (void)ws_size;
  const float* seq_pads = (const float*)d_in[0];
  const int*   seq_lens = (const int*)  d_in[1];
  const float* Wr = (const float*)d_in[2];
  const float* Ur = (const float*)d_in[3];
  const float* br = (const float*)d_in[4];
  const float* Wq = (const float*)d_in[5];
  const float* Uq = (const float*)d_in[6];
  const float* bq = (const float*)d_in[7];
  const float* Ws = (const float*)d_in[8];
  const float* Us = (const float*)d_in[9];
  const float* bs = (const float*)d_in[10];

  float* out = (float*)d_out;
  const size_t blk = (size_t)M_ * B_ * LP1 * H_;   // one [M,B,L+1,H] tensor
  float* befores = out;
  float* afters  = out + blk;
  float* delta   = out + 2 * blk;                  // [B, L+1]

  build_delta_kernel<<<B_, 256, 0, stream>>>(seq_pads, seq_lens, delta);
  ctgru_scan_kernel<<<B_ / BT, NT, 0, stream>>>(Wr, Ur, br, Wq, Uq, bq,
                                                Ws, Us, bs, delta, afters);
  const size_t total4 = blk / 4;
  const int ethreads = 256;
  const int eblocks = (int)((total4 + ethreads - 1) / ethreads);
  befores_kernel<<<eblocks, ethreads, 0, stream>>>(afters, delta, befores);
}
